// Cate2Classifier_13761075216780
// MI455X (gfx1250) — compile-verified, matrix operand order checked by reading, not run on hardware
//
#include <hip/hip_runtime.h>
#include <math.h>

// ---------------------------------------------------------------------------
// MI455X (gfx1250) implementation. Conv GLU branches are the hot path
// (~0.77 TFLOP as GEMMs) -> lowered to v_wmma_f32_16x16x32_f16 with f32
// accumulate. Embedding rows are gathered into LDS by the Tensor Data Mover
// in gather mode (row indices = token ids; index 0 == zero padding row),
// completion tracked with TENSORcnt. Weights reordered K-major once so every
// WMMA fragment is two contiguous b128 loads.
// ---------------------------------------------------------------------------

#define D_EMB   512
#define C4D     2048
#define EPS_BN  1e-5f
#define ROWP    520      // f16 LDS row stride (512 + skew pad)
#define NROWS   20       // 16 output rows + 4 halo rows staged per L-tile

typedef _Float16 v16h __attribute__((ext_vector_type(16)));
typedef _Float16 h8   __attribute__((ext_vector_type(8)));
typedef float    v8f  __attribute__((ext_vector_type(8)));
typedef unsigned int u32x4 __attribute__((ext_vector_type(4)));
typedef int          i32x8 __attribute__((ext_vector_type(8)));
typedef int          i32x4 __attribute__((ext_vector_type(4)));

// Build a 16-bit A/B WMMA fragment for this lane from a K-major row.
// Lane group g = lane>>4: elements 0..7 -> K = 8g+e, 8..15 -> K = 16+8g+(e-8).
__device__ __forceinline__ v16h ld_frag(const _Float16* p, int g) {
  union { v16h v; h8 h[2]; } u;
  u.h[0] = *(const h8*)(p + 8 * g);
  u.h[1] = *(const h8*)(p + 8 * g + 16);
  return u.v;
}

// TDM gather-mode descriptor: pull `nrows` rows (32-bit indices idx8[0..7])
// of 512 f32 from `emb` into LDS at byte offset lds_addr. See CDNA5 ISA ch.8.
// This toolchain exposes the 6-arg builtin:
//   (u32x4 g0, i32x8 g1, i32x4 g2, i32x4 g3, i32x8 extra, i32 cpol)
__device__ __forceinline__ void tdm_gather_rows(unsigned int lds_addr,
                                                const float* emb,
                                                const int* idx8, int nrows) {
  const unsigned long long ga = (unsigned long long)(uintptr_t)emb;
  u32x4 g0;
  g0.x = 0xC0000001u;                 // count=1 | gather_mode | 32-bit indices
  g0.y = lds_addr;                    // LDS dest (byte address)
  g0.z = (unsigned int)ga;            // global_addr[31:0]
  g0.w = (unsigned int)((ga >> 32) & 0x01FFFFFFu) | (2u << 30);  // [56:32]|type=2
  i32x8 g1;
  g1[0] = 0x00020000;                 // workgroup_mask=0, data_size=2 (4B)
  g1[1] = (int)(512u << 16);          // tensor_dim0 = 512 (bits 79:48 lo half)
  g1[2] = (int)(0x86A0u << 16);       // tensor_dim1 = 100000 (lo16)
  g1[3] = (int)((512u << 16) | 1u);   // tensor_dim1 hi16=1, tile_dim0=512
  g1[4] = nrows;                      // tile_dim1 = #valid gather indices
  g1[5] = 512;                        // tensor_dim0_stride = 512 elements
  g1[6] = 0;
  g1[7] = 0;
  i32x4 g2, g3;
  g2[0] = idx8[0]; g2[1] = idx8[1]; g2[2] = idx8[2]; g2[3] = idx8[3];
  g3[0] = idx8[4]; g3[1] = idx8[5]; g3[2] = idx8[6]; g3[3] = idx8[7];
  i32x8 gx = {0, 0, 0, 0, 0, 0, 0, 0};
  __builtin_amdgcn_tensor_load_to_lds(g0, g1, g2, g3, gx, 0);
}

__device__ __forceinline__ int tok_at(const int* tokens, int b, int L, int l) {
  // out-of-range -> row 0 of the embedding table, which is the zero padding
  // row (padding_idx=0), i.e. exactly the conv 'same' zero padding.
  return (l >= 0 && l < L) ? tokens[(size_t)b * L + l] : 0;
}

// ---------------------------------------------------------------------------
// Weight prep kernels
// ---------------------------------------------------------------------------

// conv w (2D, D, k) f32 -> W'[c][t*512 + d] f16 (K-major, d contiguous)
__global__ void reorder_conv_w(const float* __restrict__ w,
                               _Float16* __restrict__ wr, int ktaps) {
  const int Ktot = ktaps * D_EMB;
  const int total = 1024 * Ktot;
  int idx = blockIdx.x * blockDim.x + threadIdx.x;
  if (idx >= total) return;
  int c  = idx / Ktot;
  int kk = idx - c * Ktot;
  int t  = kk >> 9;
  int d  = kk & 511;
  wr[idx] = (_Float16)w[(size_t)c * (D_EMB * ktaps) + (size_t)d * ktaps + t];
}

__global__ void cvt_f32_f16(const float* __restrict__ s,
                            _Float16* __restrict__ d, int n) {
  int idx = blockIdx.x * blockDim.x + threadIdx.x;
  if (idx < n) d[idx] = (_Float16)s[idx];
}

// clf_w (135,1024) f32 -> (144,1024) f16, zero padded rows
__global__ void cvt_clf(const float* __restrict__ s, _Float16* __restrict__ d) {
  int idx = blockIdx.x * blockDim.x + threadIdx.x;
  if (idx >= 144 * 1024) return;
  int n = idx >> 10;
  int k = idx & 1023;
  d[idx] = (n < 135) ? (_Float16)s[n * 1024 + k] : (_Float16)0.f;
}

// ---------------------------------------------------------------------------
// Fused embedding-gather (TDM) + conv1d(k) + GLU + max-over-L.
// One (b, 128-chan) slab per block, 8 waves; each wave computes the paired
// a/g 16x16 tiles so the gated product and L-max stay in registers.
// joff = 2 - pad maps output row m, tap t to staged LDS row j = m + t + joff;
// staged row j <-> source position l0 + j - 2 (token 0 => zeros outside L).
// ---------------------------------------------------------------------------
__global__ __launch_bounds__(256)
void glu_conv_wmma(const int* __restrict__ tokens, int L, int n_lt,
                   const float* __restrict__ emb,
                   const _Float16* __restrict__ Wr,
                   const float* __restrict__ cbias,
                   int Ktot, int joff, float* __restrict__ outp) {
  __shared__ float    s_f32[NROWS * D_EMB];   // raw TDM landing buffer (40KB)
  __shared__ _Float16 s_f16[NROWS * ROWP];    // converted, skewed (20.3KB)

  const int tid  = threadIdx.x;
  const int lane = tid & 31;
  const int wv   = tid >> 5;
  const int b    = blockIdx.y;
  const int c0   = blockIdx.x * 128 + wv * 16;
  const int i15  = lane & 15;        // A: row M, B: col N, epilogue: col N
  const int g    = lane >> 4;

  const _Float16* brow_a = Wr + (size_t)(c0 + i15) * Ktot;
  const _Float16* brow_g = Wr + (size_t)(c0 + i15 + 512) * Ktot;
  const float bias_a = cbias[c0 + i15];
  const float bias_g = cbias[512 + c0 + i15];
  const int nk = Ktot >> 5;

  float maxv = -INFINITY;

  for (int lt = 0; lt < n_lt; ++lt) {
    const int l0 = lt << 4;
    __syncthreads();  // previous tile's LDS reads done before TDM overwrites

    if (wv == 0) {
      // gather the 20 rows of this tile via 3 TDM gather descriptors
      int idx[24];
#pragma unroll
      for (int j = 0; j < 20; ++j) idx[j] = tok_at(tokens, b, L, l0 + j - 2);
      idx[20] = idx[21] = idx[22] = idx[23] = 0;
      const unsigned int lb = (unsigned int)(uintptr_t)(void*)s_f32;
      tdm_gather_rows(lb,                     emb, idx + 0, 8);
      tdm_gather_rows(lb + 8 * 2048,          emb, idx + 8, 8);
      tdm_gather_rows(lb + 16 * 2048,         emb, idx + 16, 4);
      __builtin_amdgcn_s_wait_tensorcnt(0);
    }
    __syncthreads();

    // packed convert f32 -> f16 into the skewed WMMA staging buffer
    for (int cid = tid; cid < NROWS * 128; cid += 256) {
      const float4 f = ((const float4*)s_f32)[cid];
      const int j  = cid >> 7;
      const int c4 = cid & 127;
      _Float16* dst = &s_f16[j * ROWP + (c4 << 2)];
      dst[0] = (_Float16)f.x; dst[1] = (_Float16)f.y;
      dst[2] = (_Float16)f.z; dst[3] = (_Float16)f.w;
    }
    __syncthreads();

    v8f acc_a = {};
    v8f acc_g = {};
    for (int ks = 0; ks < nk; ++ks) {
      const int kk0 = ks << 5;
      const int t   = kk0 >> 9;   // K = t*512 + d  (t = conv tap)
      const int d0  = kk0 & 511;
      v16h af  = ld_frag(&s_f16[(i15 + t + joff) * ROWP + d0], g);
      v16h bfa = ld_frag(brow_a + kk0, g);
      v16h bfg = ld_frag(brow_g + kk0, g);
      acc_a = __builtin_amdgcn_wmma_f32_16x16x32_f16(
          false, af, false, bfa, (short)0, acc_a, false, false);
      acc_g = __builtin_amdgcn_wmma_f32_16x16x32_f16(
          false, af, false, bfg, (short)0, acc_g, false, false);
    }

    // GLU + masked max over this tile's rows (C/D layout: lane holds
    // col = lane&15, rows M = r + 8*(lane>>4), r = 0..7)
#pragma unroll
    for (int r = 0; r < 8; ++r) {
      const int l = l0 + r + 8 * g;
      if (l < L) {
        const float va = acc_a[r] + bias_a;
        const float vg = acc_g[r] + bias_g;
        const float v  = va * (1.0f / (1.0f + __expf(-vg)));
        maxv = fmaxf(maxv, v);
      }
    }
  }

  // merge the two row-halves (lane <-> lane^16 hold same column)
  const float oth = __shfl_xor(maxv, 16, 32);
  maxv = fmaxf(maxv, oth);
  if (lane < 16) outp[(size_t)b * C4D + c0 + lane] = maxv;
}

// ---------------------------------------------------------------------------
// Generic WMMA GEMM: C[M,ldc] = A[M,K](f16) * Bw[N,K](f16)^T + bias
// grid = (M/16, ceil(Npad/128)), block = 256 (8 waves, one 16x16 tile each).
// ---------------------------------------------------------------------------
__global__ __launch_bounds__(256)
void gemm_wmma_bias(const _Float16* __restrict__ A,
                    const _Float16* __restrict__ Bw,
                    const float* __restrict__ bias, float* __restrict__ C,
                    int K, int Npad, int Nvalid, int ldc) {
  const int tid  = threadIdx.x;
  const int lane = tid & 31;
  const int wv   = tid >> 5;
  const int m0   = blockIdx.x << 4;
  const int n0   = blockIdx.y * 128 + wv * 16;
  if (n0 >= Npad) return;            // wave-uniform guard (EXEC stays all-1)
  const int i15 = lane & 15;
  const int g   = lane >> 4;

  const _Float16* arow = A + (size_t)(m0 + i15) * K;
  const _Float16* brow = Bw + (size_t)(n0 + i15) * K;
  v8f acc = {};
  const int nk = K >> 5;
  for (int ks = 0; ks < nk; ++ks) {
    const int kk0 = ks << 5;
    v16h af = ld_frag(arow + kk0, g);
    v16h bf = ld_frag(brow + kk0, g);
    acc = __builtin_amdgcn_wmma_f32_16x16x32_f16(
        false, af, false, bf, (short)0, acc, false, false);
  }
  const int col = n0 + i15;
  if (col < Nvalid) {
    const float bb = bias[col];
#pragma unroll
    for (int r = 0; r < 8; ++r)
      C[(size_t)(m0 + r + 8 * g) * ldc + col] = acc[r] + bb;
  }
}

// ---------------------------------------------------------------------------
// Train-mode BatchNorm over batch (B=256), one block per column, f16 output
// (feeds the next WMMA GEMM). Optional ReLU.
// ---------------------------------------------------------------------------
__global__ __launch_bounds__(256)
void bn_train(const float* __restrict__ X, const float* __restrict__ gm,
              const float* __restrict__ bt, _Float16* __restrict__ Y,
              int C, int relu) {
  __shared__ float s1[256], s2[256];
  const int col = blockIdx.x;
  const int tid = threadIdx.x;
  const float v = X[(size_t)tid * C + col];
  s1[tid] = v;
  s2[tid] = v * v;
  __syncthreads();
  for (int s = 128; s > 0; s >>= 1) {
    if (tid < s) { s1[tid] += s1[tid + s]; s2[tid] += s2[tid + s]; }
    __syncthreads();
  }
  const float mean = s1[0] * (1.0f / 256.0f);
  const float var  = s2[0] * (1.0f / 256.0f) - mean * mean;
  float xn = (v - mean) * rsqrtf(var + EPS_BN) * gm[col] + bt[col];
  if (relu) xn = fmaxf(xn, 0.0f);
  Y[(size_t)tid * C + col] = (_Float16)xn;
}

// output[i, mask1[cate1[i]]] = -100
__global__ void scatter_mask(const int* __restrict__ cate1,
                             const int* __restrict__ mask1,
                             float* __restrict__ out) {
  int idx = blockIdx.x * blockDim.x + threadIdx.x;
  if (idx >= 256 * 20) return;
  int i = idx / 20;
  int m = idx - i * 20;
  int col = mask1[cate1[i] * 20 + m];
  out[(size_t)i * 135 + col] = -100.0f;
}

// ---------------------------------------------------------------------------
extern "C" void kernel_launch(void* const* d_in, const int* in_sizes, int n_in,
                              void* d_out, int out_size, void* d_ws,
                              size_t ws_size, hipStream_t stream) {
  (void)in_sizes; (void)n_in; (void)out_size; (void)ws_size;

  const int*   title   = (const int*)  d_in[0];
  const int*   desc    = (const int*)  d_in[1];
  const int*   cate1   = (const int*)  d_in[2];
  const int*   mask1   = (const int*)  d_in[3];
  const float* emb     = (const float*)d_in[4];
  const float* conv3_w = (const float*)d_in[5];
  const float* conv3_b = (const float*)d_in[6];
  const float* conv5_w = (const float*)d_in[7];
  const float* conv5_b = (const float*)d_in[8];
  const float* fc_w    = (const float*)d_in[9];
  const float* fc_b    = (const float*)d_in[10];
  const float* clf_w   = (const float*)d_in[11];
  const float* clf_b   = (const float*)d_in[12];
  const float* bn1_g   = (const float*)d_in[13];
  const float* bn1_b   = (const float*)d_in[14];
  const float* bn2_g   = (const float*)d_in[15];
  const float* bn2_b   = (const float*)d_in[16];
  float* out = (float*)d_out;

  // Workspace layout (all offsets 256B-aligned, total ~17.6 MB)
  char* ws = (char*)d_ws;
  _Float16* W3r  = (_Float16*)(ws + 0);          // 1024*1536*2 = 3,145,728
  _Float16* W5r  = (_Float16*)(ws + 3145728);    // 1024*2560*2 = 5,242,880
  _Float16* fcw  = (_Float16*)(ws + 8388608);    // 1024*2048*2 = 4,194,304
  _Float16* clfw = (_Float16*)(ws + 12582912);   // 144*1024*2  =   294,912
  float*    glu  = (float*)   (ws + 12877824);   // 256*2048*4  = 2,097,152
  _Float16* xn   = (_Float16*)(ws + 14974976);   // 256*2048*2  = 1,048,576
  float*    yfc  = (float*)   (ws + 16023552);   // 256*1024*4  = 1,048,576
  _Float16* hn   = (_Float16*)(ws + 17072128);   // 256*1024*2  =   524,288

  // --- weight prep (K-major reorder + f16 convert) ---
  reorder_conv_w<<<(1024 * 1536 + 255) / 256, 256, 0, stream>>>(conv3_w, W3r, 3);
  reorder_conv_w<<<(1024 * 2560 + 255) / 256, 256, 0, stream>>>(conv5_w, W5r, 5);
  cvt_f32_f16<<<(1024 * 2048 + 255) / 256, 256, 0, stream>>>(fc_w, fcw, 1024 * 2048);
  cvt_clf<<<(144 * 1024 + 255) / 256, 256, 0, stream>>>(clf_w, clfw);

  // --- fused gather + conv GLU + maxpool, slices [t1 | d1 | t2 | d2] ---
  dim3 gglu(4, 256);
  glu_conv_wmma<<<gglu, 256, 0, stream>>>(title,  60,  4, emb, W3r, conv3_b, 1536, 1, glu + 0);
  glu_conv_wmma<<<gglu, 256, 0, stream>>>(desc,  300, 19, emb, W3r, conv3_b, 1536, 1, glu + 512);
  glu_conv_wmma<<<gglu, 256, 0, stream>>>(title,  60,  4, emb, W5r, conv5_b, 2560, 0, glu + 1024);
  glu_conv_wmma<<<gglu, 256, 0, stream>>>(desc,  300, 19, emb, W5r, conv5_b, 2560, 0, glu + 1536);

  // --- BN1 -> FC -> BN2+ReLU -> classifier -> scatter ---
  bn_train<<<2048, 256, 0, stream>>>(glu, bn1_g, bn1_b, xn, 2048, 0);
  gemm_wmma_bias<<<dim3(16, 8), 256, 0, stream>>>(xn, fcw, fc_b, yfc, 2048, 1024, 1024, 1024);
  bn_train<<<1024, 256, 0, stream>>>(yfc, bn2_g, bn2_b, hn, 1024, 1);
  gemm_wmma_bias<<<dim3(16, 2), 256, 0, stream>>>(hn, clfw, clf_b, out, 1024, 144, 135, 135);
  scatter_mask<<<(256 * 20 + 255) / 256, 256, 0, stream>>>(cate1, mask1, out);
}
